// EvidentialGraphLearner_2954937500248
// MI455X (gfx1250) — matrix operand order repaired and checked by reading, not compiled, not run on hardware
//
#include <hip/hip_runtime.h>
#include <hip/hip_bf16.h>

// ---------------- constants from the reference ----------------
#define BB    4
#define NN    2048
#define DD    128
#define KK    32          // TOP_K
#define H1    128         // 2*D_PAIR
#define DP    64          // D_PAIR
#define NEV   4           // N_EVIDENCE

typedef __attribute__((ext_vector_type(16))) _Float16 v16h;
typedef __attribute__((ext_vector_type(8)))  _Float16 vh8;
typedef __attribute__((ext_vector_type(4)))  _Float16 vh4;
typedef __attribute__((ext_vector_type(8)))  float    v8f;
typedef __attribute__((ext_vector_type(4)))  float    v4f;

__device__ __forceinline__ float geluf(float x) {
    return 0.5f * x * (1.0f + erff(x * 0.70710678118654752f));
}
__device__ __forceinline__ float softplusf(float x) {
    return (x > 20.0f) ? x : log1pf(expf(x));
}
__device__ __forceinline__ v16h mk16(vh8 lo, vh8 hi) {
    return __builtin_shufflevector(lo, hi, 0,1,2,3,4,5,6,7,8,9,10,11,12,13,14,15);
}

// ---------------------------------------------------------------
// Kernel 0: pack an f32 weight matrix [K x ncols] into WMMA
// B-fragment order, f16: frag(c,t) -> 32 lanes x 16 halfs, so a
// lane's fragment is one contiguous 32B load.
//   kr  = 32c + 8*(lane>>4) + (q&7) + 16*(q>>3)  (B layout: lane=col, K=8*hi+(q&7)+16*(q>>3))
//   col = 16t + (lane&15)
// ---------------------------------------------------------------
__global__ void packw_kernel(const float* __restrict__ src, _Float16* __restrict__ dst,
                             int nChunks, int nTiles, int ncols) {
    const int total = nChunks * nTiles * 512;
    for (int e = blockIdx.x * blockDim.x + threadIdx.x; e < total;
         e += gridDim.x * blockDim.x) {
        const int frag = e >> 9;
        const int lane = (e >> 4) & 31;
        const int q    = e & 15;
        const int c    = frag / nTiles;
        const int t    = frag - c * nTiles;
        const int kr   = 32 * c + 8 * (lane >> 4) + (q & 7) + 16 * (q >> 3);
        const int col  = 16 * t + (lane & 15);
        dst[e] = (_Float16)src[kr * ncols + col];
    }
}

// ---------------------------------------------------------------
// Kernel 1: exact top-k (k=32) per A_prior row, stable tie-break.
// ---------------------------------------------------------------
__global__ __launch_bounds__(256)
void topk_kernel(const float* __restrict__ A, int* __restrict__ nbr,
                 float* __restrict__ apv) {
    __shared__ float vals[NN];
    __shared__ float rv[256];
    __shared__ int   ri[256];
    const int row = blockIdx.x;               // b*N + i
    const float* arow = A + (long long)row * NN;
    const int t = threadIdx.x;
    for (int j = t; j < NN; j += 256) vals[j] = arow[j];
    __syncthreads();
    for (int sel = 0; sel < KK; ++sel) {
        float bv = -3.4e38f; int bi = 0x7fffffff;
        for (int j = t; j < NN; j += 256) {
            float v = vals[j];
            if (v > bv) { bv = v; bi = j; }   // ascending visit => first occurrence kept
        }
        rv[t] = bv; ri[t] = bi;
        __syncthreads();
        for (int s = 128; s > 0; s >>= 1) {
            if (t < s) {
                float ov = rv[t + s]; int oi = ri[t + s];
                if (ov > rv[t] || (ov == rv[t] && oi < ri[t])) { rv[t] = ov; ri[t] = oi; }
            }
            __syncthreads();
        }
        if (t == 0) {
            nbr[(long long)row * KK + sel] = ri[0];
            apv[(long long)row * KK + sel] = rv[0];
            vals[ri[0]] = -3.4e38f;
        }
        __syncthreads();
    }
}

// ---------------------------------------------------------------
// Kernel 2: zero-init dense scatter target G (ws) and uncertainty.
// ---------------------------------------------------------------
__global__ void zero_kernel(float* __restrict__ G, float* __restrict__ U, long long n) {
    long long i = (long long)blockIdx.x * blockDim.x + threadIdx.x;
    const long long stride = (long long)gridDim.x * blockDim.x;
    for (; i < n; i += stride) { G[i] = 0.0f; U[i] = 0.0f; }
}

// ---------------------------------------------------------------
// Kernel 3: WMMA pair-MLP + evidential heads + sparse scatter.
// 128 threads = 4 waves; wave owns 16 rows (half of one node's k=32
// rows); block covers 2 nodes.
// A-fragments come from LDS-staged f16 rows (two 16B ds loads each),
// B-fragments from the pre-packed f16 arrays (one 32B global load).
// ---------------------------------------------------------------
__global__ __launch_bounds__(128)
void mlp_kernel(const float* __restrict__ h,   const float* __restrict__ pos,
                const float* __restrict__ W1,  const float* __restrict__ b1,
                const float* __restrict__ b2,
                const float* __restrict__ bm1, const float* __restrict__ Wm2,
                const float* __restrict__ bm2, const float* __restrict__ ba1,
                const float* __restrict__ Wa2, const float* __restrict__ ba2,
                const float* __restrict__ mu_scale_p,
                const _Float16* __restrict__ W1pk, const _Float16* __restrict__ W2pk,
                const _Float16* __restrict__ Wm1pk, const _Float16* __restrict__ Wa1pk,
                const int* __restrict__ nbr,   const float* __restrict__ apv,
                float* __restrict__ G, float* __restrict__ Uout,
                float* __restrict__ alpha_out)
{
    // per-wave scratch: 4352B buffer reused as
    //   phase 1: sHj  [16][136] f16  (neighbor rows, f16)
    //   phase 2: sH   [16][136] f16  (GEMM1 hidden, f16)
    //   phase 3: sT   [16][68]  f32  (head staging)
    __shared__ __align__(16) unsigned char sBuf[4][4352];
    __shared__ __align__(16) _Float16 sHiA[4][136];    // h_i row, f16
    __shared__ __align__(16) _Float16 sPF[4][16 * 72]; // pair_feat f16
    __shared__ float sDp[4][16][4];
    __shared__ int   sJ[4][16];

    const int tid  = threadIdx.x;
    const int lane = tid & 31;
    const int wv   = tid >> 5;
    const int rr   = lane & 15;       // A-row / B,C-col index within tile
    const int hiL  = lane >> 4;

    const int g  = blockIdx.x * 2 + (wv >> 1);   // node index in [0, B*N)
    const int b  = g >> 11;                       // N = 2048
    const int i  = g & (NN - 1);
    const int kk = ((wv & 1) << 4) + rr;
    const long long R = (long long)g * KK + kk;

    _Float16* sHj = (_Float16*)sBuf[wv];
    _Float16* sH  = (_Float16*)sBuf[wv];
    float*    sT  = (float*)sBuf[wv];
    _Float16* sHi = sHiA[wv];
    _Float16* sPFw = sPF[wv];

    const float* hp_i = h + ((long long)b * NN + i) * DD;

    // ---- stage per-row metadata + h_i (f32 -> f16) ----
    if (lane < 16) {
        const int j = nbr[R];
        sJ[wv][rr] = j;
        sDp[wv][rr][0] = pos[i * 3 + 0] - pos[j * 3 + 0];
        sDp[wv][rr][1] = pos[i * 3 + 1] - pos[j * 3 + 1];
        sDp[wv][rr][2] = pos[i * 3 + 2] - pos[j * 3 + 2];
    }
    {
        v4f x = *(const v4f*)(hp_i + 4 * lane);
        vh4 hx; hx[0] = (_Float16)x[0]; hx[1] = (_Float16)x[1];
        hx[2] = (_Float16)x[2]; hx[3] = (_Float16)x[3];
        *(vh4*)&sHi[4 * lane] = hx;
    }
    __syncthreads();

    // ---- stage 16 neighbor rows h_j (f32 -> f16), cooperative ----
    for (int idx = lane; idx < 16 * 32; idx += 32) {
        const int m  = idx >> 5;          // row
        const int q4 = idx & 31;          // float4 index within row
        const float* hp_j = h + ((long long)b * NN + sJ[wv][m]) * DD;
        v4f x = *(const v4f*)(hp_j + 4 * q4);
        vh4 hx; hx[0] = (_Float16)x[0]; hx[1] = (_Float16)x[1];
        hx[2] = (_Float16)x[2]; hx[3] = (_Float16)x[3];
        *(vh4*)&sHj[m * 136 + 4 * q4] = hx;
    }
    __syncthreads();

    const v8f z8 = {0.f, 0.f, 0.f, 0.f, 0.f, 0.f, 0.f, 0.f};
    const v16h* W1f  = (const v16h*)W1pk;
    const v16h* W2f  = (const v16h*)W2pk;
    const v16h* Wm1f = (const v16h*)Wm1pk;
    const v16h* Wa1f = (const v16h*)Wa1pk;

    // ---- GEMM1: [16 x 384] @ W1[0:384] -> hidden [16 x 128] ----
    v8f acc[8];
#pragma unroll
    for (int t = 0; t < 8; ++t) acc[t] = z8;

    const _Float16* sHjr = sHj + rr * 136;
#pragma unroll
    for (int c = 0; c < 12; ++c) {
        const int seg = c >> 2;              // 0: h_i, 1: h_j, 2: h_i*h_j
        const int off0 = (c & 3) * 32 + hiL * 8;
        v16h a;
        if (seg == 0) {
            vh8 lo = *(const vh8*)&sHi[off0];
            vh8 hi = *(const vh8*)&sHi[off0 + 16];
            a = mk16(lo, hi);
        } else if (seg == 1) {
            vh8 lo = *(const vh8*)&sHjr[off0];
            vh8 hi = *(const vh8*)&sHjr[off0 + 16];
            a = mk16(lo, hi);
        } else {
            vh8 li = *(const vh8*)&sHi[off0];
            vh8 hi2 = *(const vh8*)&sHi[off0 + 16];
            vh8 lj = *(const vh8*)&sHjr[off0];
            vh8 hj2 = *(const vh8*)&sHjr[off0 + 16];
            a = mk16(li * lj, hi2 * hj2);
        }
#pragma unroll
        for (int t = 0; t < 8; ++t) {
            v16h bf = W1f[(c * 8 + t) * 32 + lane];
            acc[t] = __builtin_amdgcn_wmma_f32_16x16x32_f16(
                false, a, false, bf, (short)0, acc[t], false, false);
        }
    }

    // epilogue: + delta_pos @ W1[384:387] + b1, GELU, stage f16
    // (sHj is dead here; sH reuses the same buffer)
#pragma unroll
    for (int t = 0; t < 8; ++t) {
        const int col = t * 16 + rr;
        const float w0 = W1[384 * H1 + col], w1 = W1[385 * H1 + col], w2 = W1[386 * H1 + col];
        const float bb1 = b1[col];
#pragma unroll
        for (int v = 0; v < 8; ++v) {
            const int M = v + 8 * hiL;
            float val = acc[t][v]
                      + sDp[wv][M][0] * w0 + sDp[wv][M][1] * w1 + sDp[wv][M][2] * w2 + bb1;
            sH[M * 136 + col] = (_Float16)geluf(val);
        }
    }
    __syncthreads();

    // ---- GEMM2: hidden [16 x 128] @ W2 -> pair_feat [16 x 64] ----
    v8f pfa[4];
#pragma unroll
    for (int t = 0; t < 4; ++t) pfa[t] = z8;
#pragma unroll
    for (int c = 0; c < 4; ++c) {
        const int off0 = rr * 136 + c * 32 + hiL * 8;
        vh8 lo = *(const vh8*)&sH[off0];
        vh8 hi = *(const vh8*)&sH[off0 + 16];
        v16h a = mk16(lo, hi);
#pragma unroll
        for (int t = 0; t < 4; ++t) {
            v16h bf = W2f[(c * 4 + t) * 32 + lane];
            pfa[t] = __builtin_amdgcn_wmma_f32_16x16x32_f16(
                false, a, false, bf, (short)0, pfa[t], false, false);
        }
    }
#pragma unroll
    for (int t = 0; t < 4; ++t) {
        const int col = t * 16 + rr;
        const float bb2 = b2[col];
#pragma unroll
        for (int v = 0; v < 8; ++v) {
            const int M = v + 8 * hiL;
            sPFw[M * 72 + col] = (_Float16)(pfa[t][v] + bb2);
        }
    }
    __syncthreads();

    // ---- head 1 (mu): GELU(pf @ Wm1 + bm1) -> sT, dot with Wm2 ----
    v8f ha[4];
#pragma unroll
    for (int t = 0; t < 4; ++t) ha[t] = z8;
#pragma unroll
    for (int c = 0; c < 2; ++c) {
        const int off0 = rr * 72 + c * 32 + hiL * 8;
        vh8 lo = *(const vh8*)&sPFw[off0];
        vh8 hi = *(const vh8*)&sPFw[off0 + 16];
        v16h a = mk16(lo, hi);
#pragma unroll
        for (int t = 0; t < 4; ++t) {
            v16h bf = Wm1f[(c * 4 + t) * 32 + lane];
            ha[t] = __builtin_amdgcn_wmma_f32_16x16x32_f16(
                false, a, false, bf, (short)0, ha[t], false, false);
        }
    }
#pragma unroll
    for (int t = 0; t < 4; ++t) {
        const int col = t * 16 + rr;
        const float bbm = bm1[col];
#pragma unroll
        for (int v = 0; v < 8; ++v) {
            const int M = v + 8 * hiL;
            sT[M * 68 + col] = geluf(ha[t][v] + bbm);
        }
    }
    __syncthreads();

    float mu_raw = 0.0f;
    if (lane < 16) {
        mu_raw = bm2[0];
        for (int c = 0; c < DP; ++c) mu_raw += sT[rr * 68 + c] * Wm2[c];
    }
    __syncthreads();   // sT about to be overwritten by head 2

    // ---- head 2 (alpha): GELU(pf @ Wa1 + ba1) -> sT ----
#pragma unroll
    for (int t = 0; t < 4; ++t) ha[t] = z8;
#pragma unroll
    for (int c = 0; c < 2; ++c) {
        const int off0 = rr * 72 + c * 32 + hiL * 8;
        vh8 lo = *(const vh8*)&sPFw[off0];
        vh8 hi = *(const vh8*)&sPFw[off0 + 16];
        v16h a = mk16(lo, hi);
#pragma unroll
        for (int t = 0; t < 4; ++t) {
            v16h bf = Wa1f[(c * 4 + t) * 32 + lane];
            ha[t] = __builtin_amdgcn_wmma_f32_16x16x32_f16(
                false, a, false, bf, (short)0, ha[t], false, false);
        }
    }
#pragma unroll
    for (int t = 0; t < 4; ++t) {
        const int col = t * 16 + rr;
        const float bba = ba1[col];
#pragma unroll
        for (int v = 0; v < 8; ++v) {
            const int M = v + 8 * hiL;
            sT[M * 68 + col] = geluf(ha[t][v] + bba);
        }
    }
    __syncthreads();

    // ---- per-row tail: alpha, u, mu, gated scatter ----
    if (lane < 16) {
        float S = 0.0f;
#pragma unroll
        for (int e = 0; e < NEV; ++e) {
            float x = ba2[e];
            for (int c = 0; c < DP; ++c) x += sT[rr * 68 + c] * Wa2[c * NEV + e];
            x = softplusf(x) + 1.0f;
            x = fminf(fmaxf(x, 1.01f), 1000.0f);
            S += x;
            alpha_out[R * NEV + e] = x;
        }
        const float u   = fminf((float)NEV / S, 0.999f);
        const float msp = softplusf(mu_scale_p[0]);
        const float mu  = tanhf(mu_raw) * msp;
        const float Ag  = (apv[R] + mu) * (1.0f - u);
        const int jn = sJ[wv][rr];
        const long long base = ((long long)b * NN + i) * NN + jn;
        G[base]    = Ag;   // top-k indices unique per row -> no atomics
        Uout[base] = u;
    }
}

// ---------------------------------------------------------------
// Kernel 4: row sums of relu(0.5*(G + G^T)), clamped at 1e-8.
// Transposed column walk gets explicit prefetch (global_prefetch_b8).
// ---------------------------------------------------------------
__global__ __launch_bounds__(256)
void rowsum_kernel(const float* __restrict__ G, float* __restrict__ rs) {
    const int row = blockIdx.x;           // b*N + i
    const int b = row >> 11;
    const int i = row & (NN - 1);
    const float* gr  = G + (long long)row * NN;
    const float* gcb = G + (long long)b * NN * NN + i;   // G[b, j, i] = gcb[j*NN]
    float s = 0.0f;
    for (int j = threadIdx.x; j < NN; j += 256) {
        if (j + 256 < NN) __builtin_prefetch(&gcb[(long long)(j + 256) * NN], 0, 1);
        float v = 0.5f * (gr[j] + gcb[(long long)j * NN]);
        s += fmaxf(v, 0.0f);
    }
    __shared__ float red[256];
    red[threadIdx.x] = s;
    __syncthreads();
    for (int st = 128; st > 0; st >>= 1) {
        if (threadIdx.x < st) red[threadIdx.x] += red[threadIdx.x + st];
        __syncthreads();
    }
    if (threadIdx.x == 0) rs[row] = fmaxf(red[0], 1e-8f);
}

// ---------------------------------------------------------------
// Kernel 5: A_eff = relu(0.5*(G + G^T)) / rowsum  -> d_out region 0
// ---------------------------------------------------------------
__global__ void norm_kernel(const float* __restrict__ G, const float* __restrict__ rs,
                            float* __restrict__ Aout) {
    const long long total = (long long)BB * NN * NN;
    long long idx = (long long)blockIdx.x * blockDim.x + threadIdx.x;
    const long long stride = (long long)gridDim.x * blockDim.x;
    for (; idx < total; idx += stride) {
        const long long bnn = (long long)NN * NN;
        const int b = (int)(idx / bnn);
        const long long rem = idx - (long long)b * bnn;
        const int r = (int)(rem / NN);
        const int j = (int)(rem - (long long)r * NN);
        float v = 0.5f * (G[idx] + G[(long long)b * bnn + (long long)j * NN + r]);
        v = fmaxf(v, 0.0f);
        Aout[idx] = v / rs[b * NN + r];
    }
}

// ---------------------------------------------------------------
extern "C" void kernel_launch(void* const* d_in, const int* in_sizes, int n_in,
                              void* d_out, int out_size, void* d_ws, size_t ws_size,
                              hipStream_t stream) {
    const float* h_in   = (const float*)d_in[0];
    const float* pos    = (const float*)d_in[1];
    const float* Aprior = (const float*)d_in[2];
    const float* W1  = (const float*)d_in[3];
    const float* b1  = (const float*)d_in[4];
    const float* W2  = (const float*)d_in[5];
    const float* b2  = (const float*)d_in[6];
    const float* Wm1 = (const float*)d_in[7];
    const float* bm1 = (const float*)d_in[8];
    const float* Wm2 = (const float*)d_in[9];
    const float* bm2 = (const float*)d_in[10];
    const float* Wa1 = (const float*)d_in[11];
    const float* ba1 = (const float*)d_in[12];
    const float* Wa2 = (const float*)d_in[13];
    const float* ba2 = (const float*)d_in[14];
    const float* mu_scale = (const float*)d_in[15];

    const long long BNK = (long long)BB * NN * KK;   // 262144
    const long long BNN = (long long)BB * NN * NN;   // 16777216

    // workspace layout (all offsets 32B-aligned):
    // [ nbr int | apv f32 | G f32 dense | rs f32 | W1pk | W2pk | Wm1pk | Wa1pk ]
    char* ws = (char*)d_ws;
    int*      nbr   = (int*)ws;
    float*    apv   = (float*)(ws + BNK * 4);
    float*    G     = (float*)(ws + 2 * BNK * 4);
    float*    rs    = (float*)(ws + 2 * BNK * 4 + BNN * 4);
    char*     pk    = ws + 2 * BNK * 4 + BNN * 4 + (long long)BB * NN * 4;
    _Float16* W1pk  = (_Float16*)pk;                       // 96 frags * 512 halfs
    _Float16* W2pk  = (_Float16*)(pk + 98304);             // 16 frags
    _Float16* Wm1pk = (_Float16*)(pk + 98304 + 16384);     //  8 frags
    _Float16* Wa1pk = (_Float16*)(pk + 98304 + 16384 + 8192);

    // output layout: [ A_eff | uncertainty | alpha ]
    float* Aout  = (float*)d_out;
    float* Uout  = Aout + BNN;
    float* alpha = Aout + 2 * BNN;

    const int rows = BB * NN;                 // 8192

    packw_kernel<<<192, 256, 0, stream>>>(W1,  W1pk, 12, 8, H1);
    packw_kernel<<<32,  256, 0, stream>>>(W2,  W2pk,  4, 4, DP);
    packw_kernel<<<16,  256, 0, stream>>>(Wm1, Wm1pk, 2, 4, DP);
    packw_kernel<<<16,  256, 0, stream>>>(Wa1, Wa1pk, 2, 4, DP);
    topk_kernel<<<rows, 256, 0, stream>>>(Aprior, nbr, apv);
    zero_kernel<<<4096, 256, 0, stream>>>(G, Uout, BNN);
    mlp_kernel<<<rows / 2, 128, 0, stream>>>(h_in, pos, W1, b1, b2,
                                             bm1, Wm2, bm2, ba1, Wa2, ba2, mu_scale,
                                             W1pk, W2pk, Wm1pk, Wa1pk,
                                             nbr, apv, G, Uout, alpha);
    rowsum_kernel<<<rows, 256, 0, stream>>>(G, rs);
    norm_kernel<<<8192, 256, 0, stream>>>(G, rs, Aout);
}
